// MultiHeadAttention_76416058131099
// MI455X (gfx1250) — matrix operand order repaired and checked by reading, not compiled, not run on hardware
//
#include <hip/hip_runtime.h>
#include <hip/hip_bf16.h>
#include <cstdint>
#include <cstddef>

// Problem constants (match reference)
#define B_ 4
#define N_ 2048
#define C_ 512
#define H_ 8
#define D_ 64
#define SCALE_ 0.125f   // 64^-0.5

typedef __attribute__((ext_vector_type(16))) __bf16 v16bf;
typedef __attribute__((ext_vector_type(8)))  float  v8f;

__device__ __forceinline__ __bf16 f2bf(float f) {
  union { float f; uint32_t u; } cv; cv.f = f;
  uint32_t u = cv.u;
  uint32_t r = u + 0x7FFFu + ((u >> 16) & 1u);   // round-to-nearest-even
  union { uint16_t s; __bf16 b; } o; o.s = (uint16_t)(r >> 16);
  return o.b;
}

// A-fragment (16-bit A, 16x32): lane-half hi holds one contiguous
// 16-half run: K = hi*16 + 0..15  -> two 16B loads.
__device__ __forceinline__ v16bf load_a_frag(const __bf16* rowp, int hi) {
  union { v16bf v; uint4 q[2]; } u;
  u.q[0] = *(const uint4*)(rowp + hi * 16);
  u.q[1] = *(const uint4*)(rowp + hi * 16 + 8);
  return u.v;
}

// B-fragment (16-bit B, 32x16): lane = column, halves = K = hi*16 + 0..15
// -> one contiguous 32B run = two 16B loads.
__device__ __forceinline__ v16bf load_b_frag(const __bf16* p) {
  union { v16bf v; uint4 q[2]; } u;
  u.q[0] = *(const uint4*)(p);
  u.q[1] = *(const uint4*)(p + 8);
  return u.v;
}

__device__ __forceinline__ v8f wmma_bf16(v16bf a, v16bf b, v8f c) {
  return __builtin_amdgcn_wmma_f32_16x16x32_bf16(false, a, false, b, (short)0, c,
                                                 false, false);
}

#define WT_STRIDE 136   // 128 + 8 pad (keeps 16B alignment, spreads LDS banks)

// ---------------------------------------------------------------------------
// Prep kernel A: f32 -> bf16 elementwise (x), float4 -> packed uint2
// ---------------------------------------------------------------------------
__global__ __launch_bounds__(256)
void cvt_bf16_kernel(const float* __restrict__ src, __bf16* __restrict__ dst) {
  const size_t i4 = ((size_t)blockIdx.x * 256 + threadIdx.x) * 4;
  const float4 f = *(const float4*)&src[i4];
  union { __bf16 h[4]; uint2 u; } p;
  p.h[0] = f2bf(f.x); p.h[1] = f2bf(f.y); p.h[2] = f2bf(f.z); p.h[3] = f2bf(f.w);
  *(uint2*)&dst[i4] = p.u;
}

// ---------------------------------------------------------------------------
// Prep kernel B: W [512][outd] f32 -> W^T [outd][512] bf16 (LDS 32x32 tiles,
// coalesced read and write).
// ---------------------------------------------------------------------------
__global__ __launch_bounds__(256)
void wT_bf16_kernel(const float* __restrict__ W, __bf16* __restrict__ wT, int outd) {
  __shared__ alignas(8) __bf16 t[32 * 36];
  const int tid = threadIdx.x;
  const int k0 = blockIdx.x * 32;   // input-row (k) block
  const int c0 = blockIdx.y * 32;   // output-col block
  {
    const int kl = tid >> 3, c4 = (tid & 7) << 2;
    const float4 f = *(const float4*)&W[(size_t)(k0 + kl) * outd + c0 + c4];
    t[(c4 + 0) * 36 + kl] = f2bf(f.x);
    t[(c4 + 1) * 36 + kl] = f2bf(f.y);
    t[(c4 + 2) * 36 + kl] = f2bf(f.z);
    t[(c4 + 3) * 36 + kl] = f2bf(f.w);
  }
  __syncthreads();
  {
    const int cl = tid >> 3, k4 = (tid & 7) << 2;
    *(uint2*)&wT[(size_t)(c0 + cl) * C_ + k0 + k4] = *(const uint2*)&t[cl * 36 + k4];
  }
}

// ---------------------------------------------------------------------------
// Kernel 1: qkv = xbf @ W_qkv^T(pre-bf16) + b_qkv ; scatter to Q[B,H,N,D],
//           K[B,H,N,D], V^T[B,H,D,N] bf16. Staging is pure uint2 copies.
// ---------------------------------------------------------------------------
__global__ __launch_bounds__(128)
void qkv_gemm_kernel(const __bf16* __restrict__ xbf, const __bf16* __restrict__ wqT,
                     const float* __restrict__ bq,
                     __bf16* __restrict__ qb, __bf16* __restrict__ kb,
                     __bf16* __restrict__ vTb) {
  __shared__ alignas(16) __bf16 xs[32 * 128];          // A tile (8 KB)
  __shared__ alignas(16) __bf16 wt[64 * WT_STRIDE];    // B tile [col][k] (17 KB)
  const int tid = threadIdx.x;
  const int w = tid >> 5, lane = tid & 31, ln = lane & 15, hi = lane >> 4;
  const int row0 = blockIdx.x * 32;
  const int col0 = blockIdx.y * 64;

  // Per-thread staging geometry (loop-invariant):
  const int xr0 = tid >> 5, xc4 = (tid & 31) << 2;   // xs: rows xr0+4i, cols xc4..+3
  const int wc0 = tid >> 5, wk4 = (tid & 31) << 2;   // wt: cols wc0+4i, k = wk4..+3

  v8f acc0 = {}, acc1 = {};
  for (int kk = 0; kk < C_; kk += 128) {
    // ---- phase 1: batch all global loads (24x 8B, no conversion) ----
    uint2 fx[8], fw[16];
#pragma unroll
    for (int i = 0; i < 8; ++i)
      fx[i] = *(const uint2*)&xbf[(size_t)(row0 + xr0 + i * 4) * C_ + kk + xc4];
#pragma unroll
    for (int i = 0; i < 16; ++i)
      fw[i] = *(const uint2*)&wqT[(size_t)(col0 + wc0 + i * 4) * C_ + kk + wk4];
    // ---- phase 2: LDS stores ----
#pragma unroll
    for (int i = 0; i < 8; ++i)
      *(uint2*)&xs[(xr0 + i * 4) * 128 + xc4] = fx[i];
#pragma unroll
    for (int i = 0; i < 16; ++i)
      *(uint2*)&wt[(wc0 + i * 4) * WT_STRIDE + wk4] = fw[i];
    __syncthreads();
#pragma unroll
    for (int s = 0; s < 4; ++s) {
      const v16bf b  = load_b_frag(&wt[(w * 16 + ln) * WT_STRIDE + s * 32 + hi * 16]);
      const v16bf a0 = load_a_frag(&xs[ln * 128 + s * 32], hi);
      const v16bf a1 = load_a_frag(&xs[(16 + ln) * 128 + s * 32], hi);
      acc0 = wmma_bf16(a0, b, acc0);
      acc1 = wmma_bf16(a1, b, acc1);
    }
    __syncthreads();
  }
  const int col = col0 + w * 16 + ln;
  const float bias = bq[col];
  const int s = col / C_;                // 0=q 1=k 2=v
  const int rem = col % C_;
  const int hh = rem / D_, d = rem % D_;
#pragma unroll
  for (int half = 0; half < 2; ++half) {
    const v8f acc = half ? acc1 : acc0;
#pragma unroll
    for (int r = 0; r < 8; ++r) {
      const int row = row0 + half * 16 + r + hi * 8;
      const int bb = row / N_, nn = row % N_;
      const __bf16 bv = f2bf(acc[r] + bias);
      if (s == 0)      qb[(((size_t)bb * H_ + hh) * N_ + nn) * D_ + d] = bv;
      else if (s == 1) kb[(((size_t)bb * H_ + hh) * N_ + nn) * D_ + d] = bv;
      else             vTb[(((size_t)bb * H_ + hh) * D_ + d) * N_ + nn] = bv;
    }
  }
}

// ---------------------------------------------------------------------------
// Kernel 2: fused attention per (b, 16-query tile).
//   LDS: S f32 [16][2048] (128K) + MEAN f32 [16][2048] (128K)
//        + Pbf bf16 [16][2048] (64K) = 320 KB (full CDNA5 WGP LDS)
// ---------------------------------------------------------------------------
__global__ __launch_bounds__(128)
void attn_fused_kernel(const __bf16* __restrict__ qb, const __bf16* __restrict__ kb,
                       const __bf16* __restrict__ vTb, const int* __restrict__ maskp,
                       __bf16* __restrict__ aob, float* __restrict__ out_mean) {
  extern __shared__ char smem_raw[];
  float*  S    = (float*)smem_raw;           // [16][N_]
  float*  MEAN = S + 16 * N_;                // [16][N_]
  __bf16* Pbf  = (__bf16*)(MEAN + 16 * N_);  // [16][N_]
  const int tid = threadIdx.x;
  const int w = tid >> 5, lane = tid & 31, ln = lane & 15, hi = lane >> 4;
  const int b  = blockIdx.x / (N_ / 16);
  const int q0 = (blockIdx.x % (N_ / 16)) * 16;

  for (int e = tid; e < 16 * N_; e += 128) MEAN[e] = 0.f;

  for (int h = 0; h < H_; ++h) {
    // ---- Q fragments (vector-loaded, reused across all key tiles) ----
    const __bf16* qrow = qb + (((size_t)b * H_ + h) * N_ + q0 + ln) * D_;
    const v16bf a0 = load_a_frag(qrow, hi);        // d = 0..31
    const v16bf a1 = load_a_frag(qrow + 32, hi);   // d = 32..63
    // ---- scores: S = mask(QK^T * scale), key tiles split over 4 waves ----
    const __bf16* kbase = kb + (((size_t)b * H_ + h) * N_) * D_;
#pragma unroll 2
    for (int kt = w; kt < N_ / 16; kt += 4) {
      const int key = kt * 16 + ln;
      const __bf16* krow = kbase + (size_t)key * D_;   // B-frag column = key
      const v16bf b0 = load_b_frag(krow + hi * 16);
      const v16bf b1 = load_b_frag(krow + 32 + hi * 16);
      const int mk = maskp[b * N_ + key];
      v8f acc = {};
      acc = wmma_bf16(a0, b0, acc);
      acc = wmma_bf16(a1, b1, acc);
#pragma unroll
      for (int r = 0; r < 8; ++r) {
        const int m = r + hi * 8;
        S[m * N_ + key] = mk ? acc[r] * SCALE_ : -__builtin_inff();
      }
    }
    __syncthreads();
    // ---- softmax per row (8 threads/row), write P as bf16, accumulate mean --
    {
      const int r = tid >> 3, sub = tid & 7;
      float mx = -__builtin_inff();
      for (int c = sub; c < N_; c += 8) mx = fmaxf(mx, S[r * N_ + c]);
#pragma unroll
      for (int o = 1; o < 8; o <<= 1) mx = fmaxf(mx, __shfl_xor(mx, o, 32));
      float sum = 0.f;
      for (int c = sub; c < N_; c += 8) {
        float e = __expf(S[r * N_ + c] - mx);
        S[r * N_ + c] = e;
        sum += e;
      }
#pragma unroll
      for (int o = 1; o < 8; o <<= 1) sum += __shfl_xor(sum, o, 32);
      const float inv = 1.f / sum;
      for (int c = sub; c < N_; c += 8) {
        const float p = S[r * N_ + c] * inv;
        Pbf[r * N_ + c] = f2bf(p);
        MEAN[r * N_ + c] += p * (1.0f / H_);
      }
    }
    __syncthreads();
    // ---- O = P @ V : wave w owns 16-wide d-tile; A-frag = 2x ds_load_b128 --
    {
      const int d = w * 16 + ln;
      const __bf16* vrow = vTb + (((size_t)b * H_ + h) * D_ + d) * N_;
      v8f oacc = {};
#pragma unroll 2
      for (int kc = 0; kc < N_; kc += 32) {
        const v16bf a  = load_a_frag(&Pbf[ln * N_ + kc], hi);
        const v16bf bf = load_b_frag(vrow + kc + hi * 16);
        oacc = wmma_bf16(a, bf, oacc);
      }
#pragma unroll
      for (int r = 0; r < 8; ++r) {
        const int m = r + hi * 8;
        aob[((size_t)(b * N_ + q0 + m)) * C_ + h * D_ + d] = f2bf(oacc[r]);
      }
    }
    __syncthreads();   // protect S/Pbf before next head
  }
  // ---- write attention mean for this (b, q-tile) ----
  for (int e = tid; e < 16 * N_; e += 128) {
    const int r = e / N_, c = e % N_;
    out_mean[((size_t)b * N_ + q0 + r) * N_ + c] = MEAN[e];
  }
}

// ---------------------------------------------------------------------------
// Kernel 3: out = attn_out @ W_proj^T(pre-bf16) + b_proj (f32 output)
// ---------------------------------------------------------------------------
__global__ __launch_bounds__(128)
void proj_gemm_kernel(const __bf16* __restrict__ aob, const __bf16* __restrict__ wpT,
                      const float* __restrict__ bp, float* __restrict__ out) {
  __shared__ alignas(16) __bf16 xs[32 * 128];
  __shared__ alignas(16) __bf16 wt[64 * WT_STRIDE];
  const int tid = threadIdx.x;
  const int w = tid >> 5, lane = tid & 31, ln = lane & 15, hi = lane >> 4;
  const int row0 = blockIdx.x * 32;
  const int col0 = blockIdx.y * 64;

  const int xr0 = tid >> 5, xc4 = (tid & 31) << 2;
  const int wc0 = tid >> 5, wk4 = (tid & 31) << 2;

  v8f acc0 = {}, acc1 = {};
  for (int kk = 0; kk < C_; kk += 128) {
    uint2 fa[8], fw[16];
#pragma unroll
    for (int i = 0; i < 8; ++i)
      fa[i] = *(const uint2*)&aob[(size_t)(row0 + xr0 + i * 4) * C_ + kk + xc4];
#pragma unroll
    for (int i = 0; i < 16; ++i)
      fw[i] = *(const uint2*)&wpT[(size_t)(col0 + wc0 + i * 4) * C_ + kk + wk4];
#pragma unroll
    for (int i = 0; i < 8; ++i)
      *(uint2*)&xs[(xr0 + i * 4) * 128 + xc4] = fa[i];
#pragma unroll
    for (int i = 0; i < 16; ++i)
      *(uint2*)&wt[(wc0 + i * 4) * WT_STRIDE + wk4] = fw[i];
    __syncthreads();
#pragma unroll
    for (int s = 0; s < 4; ++s) {
      const v16bf b  = load_b_frag(&wt[(w * 16 + ln) * WT_STRIDE + s * 32 + hi * 16]);
      const v16bf a0 = load_a_frag(&xs[ln * 128 + s * 32], hi);
      const v16bf a1 = load_a_frag(&xs[(16 + ln) * 128 + s * 32], hi);
      acc0 = wmma_bf16(a0, b, acc0);
      acc1 = wmma_bf16(a1, b, acc1);
    }
    __syncthreads();
  }
  const int col = col0 + w * 16 + ln;
  const float bias = bp[col];
#pragma unroll
  for (int half = 0; half < 2; ++half) {
    const v8f acc = half ? acc1 : acc0;
#pragma unroll
    for (int r = 0; r < 8; ++r) {
      const int row = row0 + half * 16 + r + hi * 8;
      out[(size_t)row * C_ + col] = acc[r] + bias;
    }
  }
}

// ---------------------------------------------------------------------------
extern "C" void kernel_launch(void* const* d_in, const int* in_sizes, int n_in,
                              void* d_out, int out_size, void* d_ws, size_t ws_size,
                              hipStream_t stream) {
  const float* x    = (const float*)d_in[0];
  const int*   mask = (const int*)d_in[1];
  const float* Wq   = (const float*)d_in[2];
  const float* bq   = (const float*)d_in[3];
  const float* Wp   = (const float*)d_in[4];
  const float* bp   = (const float*)d_in[5];

  float* out      = (float*)d_out;                       // [B,N,C]
  float* out_mean = out + (size_t)B_ * N_ * C_;          // [B,N,N]

  // Workspace layout (bf16):
  //   qb/kb/vTb/aob: 4x 4.19M elems (8 MB each)
  //   xbf: 4.19M elems (8 MB), wqT: 1536x512 (1.5 MB), wpT: 512x512 (0.5 MB)
  __bf16* qb  = (__bf16*)d_ws;
  __bf16* kb  = qb  + (size_t)B_ * H_ * N_ * D_;
  __bf16* vTb = kb  + (size_t)B_ * H_ * N_ * D_;
  __bf16* aob = vTb + (size_t)B_ * H_ * N_ * D_;
  __bf16* xbf = aob + (size_t)B_ * N_ * C_;
  __bf16* wqT = xbf + (size_t)B_ * N_ * C_;
  __bf16* wpT = wqT + (size_t)(3 * C_) * C_;

  dim3 blk128(128), blk256(256);

  // 0) one-shot precision/layout prep (memory-bound, L2-resident afterwards)
  cvt_bf16_kernel<<<dim3((B_ * N_ * C_) / (256 * 4)), blk256, 0, stream>>>(x, xbf);
  wT_bf16_kernel<<<dim3(C_ / 32, (3 * C_) / 32), blk256, 0, stream>>>(Wq, wqT, 3 * C_);
  wT_bf16_kernel<<<dim3(C_ / 32, C_ / 32), blk256, 0, stream>>>(Wp, wpT, C_);

  // 1) QKV projection: [8192,512] x [512,1536]
  qkv_gemm_kernel<<<dim3((B_ * N_) / 32, (3 * C_) / 64), blk128, 0, stream>>>(
      xbf, wqT, bq, qb, kb, vTb);

  // 2) fused attention; 320 KB dynamic LDS (full CDNA5 WGP allocation)
  const int dyn = 2 * 16 * N_ * (int)sizeof(float) + 16 * N_ * (int)sizeof(__bf16);
  hipFuncSetAttribute((const void*)attn_fused_kernel,
                      hipFuncAttributeMaxDynamicSharedMemorySize, dyn);
  attn_fused_kernel<<<dim3(B_ * (N_ / 16)), blk128, (size_t)dyn, stream>>>(
      qb, kb, vTb, mask, aob, out_mean);

  // 3) output projection: [8192,512] x [512,512]
  proj_gemm_kernel<<<dim3((B_ * N_) / 32, C_ / 64), blk128, 0, stream>>>(
      aob, wpT, bp, out);
}
